// DeformableConv2d_2594160246857
// MI455X (gfx1250) — matrix operand order repaired and checked by reading, not compiled, not run on hardware
//
#include <hip/hip_runtime.h>
#include <hip/hip_bf16.h>

typedef __attribute__((ext_vector_type(16))) _Float16 v16h;
typedef __attribute__((ext_vector_type(8)))  _Float16 v8h;
typedef __attribute__((ext_vector_type(8)))  float    v8f;

#define Bsz   4
#define Cin   128
#define Hdim  128
#define Wdim  128
#define HW    (Hdim*Wdim)    // 16384
#define COUT  128
#define KK    9
#define MAXOFF 32.0f         // max(H,W)/4
#define TILE_M 32            // pixels per block (one image row slice)

// ---------------------------------------------------------------------------
// Kernel 1: 3x3 offset conv (128 -> 27 ch), f32, weights staged in LDS.
// Emits dy/dx (clipped) and mask (sigmoid) per (b, tap, pixel).
// ---------------------------------------------------------------------------
__global__ __launch_bounds__(256) void offset_conv_kernel(
    const float* __restrict__ x, const float* __restrict__ ow,
    const float* __restrict__ ob,
    float* __restrict__ dy, float* __restrict__ dx, float* __restrict__ msk)
{
    const int tid = threadIdx.x;
    const int p   = blockIdx.x * 256 + tid;
    const int b   = p >> 14;
    const int hw  = p & (HW - 1);
    const int h   = hw >> 7;
    const int w   = hw & (Wdim - 1);

    __shared__ float sw[16 * 27 * 9];   // 16-channel slice of offset_w (15.5 KB)

    float acc[27];
    #pragma unroll
    for (int oc = 0; oc < 27; ++oc) acc[oc] = ob[oc];

    const float* xb = x + (size_t)b * Cin * HW;

    for (int cc = 0; cc < Cin / 16; ++cc) {
        __syncthreads();
        for (int i = tid; i < 16 * 27 * 9; i += 256) {
            int cl = i / 243; int r = i - cl * 243; int oc = r / 9; int t = r - oc * 9;
            sw[i] = ow[oc * (Cin * 9) + (cc * 16 + cl) * 9 + t];
        }
        __syncthreads();
        for (int cl = 0; cl < 16; ++cl) {
            const float* xc = xb + (size_t)(cc * 16 + cl) * HW;
            float xv[9];
            #pragma unroll
            for (int t = 0; t < 9; ++t) {
                int yy = h - 1 + t / 3, xx = w - 1 + t % 3;
                bool v = (yy >= 0) && (yy < Hdim) && (xx >= 0) && (xx < Wdim);
                xv[t] = v ? xc[yy * Wdim + xx] : 0.f;
            }
            const float* swc = &sw[cl * 243];
            #pragma unroll
            for (int oc = 0; oc < 27; ++oc) {
                float s = 0.f;
                #pragma unroll
                for (int t = 0; t < 9; ++t) s += xv[t] * swc[oc * 9 + t];
                acc[oc] += s;
            }
        }
    }

    // concat(o1,o2) == channels 0..17 verbatim; mask = ch 18..26
    #pragma unroll
    for (int k = 0; k < KK; ++k) {
        int idx = ((b * KK + k) << 14) + hw;
        dy[idx]  = fminf(fmaxf(acc[2 * k],     -MAXOFF), MAXOFF);
        dx[idx]  = fminf(fmaxf(acc[2 * k + 1], -MAXOFF), MAXOFF);
        msk[idx] = 1.f / (1.f + __expf(-acc[18 + k]));
    }
}

// ---------------------------------------------------------------------------
// Kernel 2: pack weight (Cout,C,KK) f32 -> per-lane f16 B fragments for
// v_wmma_f32_16x16x32_f16.  B (32x16, K x N) layout used here (contiguous K, same order as A frag):
//   lanes 0-15: N=lane,    16 halves = K 0..15
//   lanes16-31: N=lane-16, 16 halves = K 16..31
// wp[k][chunk][ntile][lane][j], one contiguous 32B read per lane per frag.
// ---------------------------------------------------------------------------
__global__ __launch_bounds__(256) void pack_weight_kernel(
    const float* __restrict__ wgt, _Float16* __restrict__ wp)
{
    int idx = blockIdx.x * 256 + threadIdx.x;
    if (idx >= KK * 4 * 8 * 32 * 16) return;
    int j     =  idx        & 15;
    int lane  = (idx >> 4)  & 31;
    int ntile = (idx >> 9)  & 7;
    int chunk = (idx >> 12) & 3;
    int k     =  idx >> 14;
    int n  = ntile * 16 + (lane & 15);
    int kl = (lane < 16) ? j : (16 + j);
    int c  = chunk * 32 + kl;
    wp[idx] = (_Float16) wgt[n * (Cin * KK) + c * KK + k];
}

// ---------------------------------------------------------------------------
// Kernel 3: fused bilinear deformable sampling + WMMA GEMM.
// Block: 256 threads (8 waves), 32 pixels x 128 Cout; wave wv owns N-tile wv
// and two M sub-tiles (B fragment reused across both -> 2 WMMAs per B load).
// Per tap: gather-bilinear 32x128 -> LDS f16, then 4 K=32 chunks x 2 M-tiles.
// ---------------------------------------------------------------------------
__global__ __launch_bounds__(256) void deform_wmma_kernel(
    const float* __restrict__ x, const _Float16* __restrict__ wp,
    const float* __restrict__ dy, const float* __restrict__ dx,
    const float* __restrict__ msk, float* __restrict__ out)
{
    const int tid  = threadIdx.x;
    const int lane = tid & 31;
    const int wv   = tid >> 5;          // 0..7 -> N-tile
    const int b    = blockIdx.x >> 9;   // 512 tiles per batch
    const int tile = blockIdx.x & 511;
    const int p_base = tile << 5;       // 32 pixels, same image row (32 | 128)
    const int h  = p_base >> 7;
    const int w0 = p_base & (Wdim - 1);

    __shared__ _Float16 As[TILE_M][Cin];  // A-tile: 32 px x 128 ch f16 (8 KB)
    __shared__ int   s_off[TILE_M][4];
    __shared__ float s_wt[TILE_M][4];
    __shared__ float s_m[TILE_M];

    v8f acc0 = {0.f, 0.f, 0.f, 0.f, 0.f, 0.f, 0.f, 0.f};
    v8f acc1 = {0.f, 0.f, 0.f, 0.f, 0.f, 0.f, 0.f, 0.f};

    const float* xb  = x + (size_t)b * Cin * HW;
    const int mrow   = lane & 15;
    const int kofs   = (lane < 16) ? 0 : 16;  // A frag: K 0-15 vs 16-31

    for (int k = 0; k < KK; ++k) {
        __syncthreads();                       // A-tile reads of prev tap done
        if (tid < TILE_M) {
            const int p   = p_base + tid;
            const int idx = ((b * KK + k) << 14) + p;
            const float py = dy[idx] + (float)(h - 1 + k / 3);
            const float px = dx[idx] + (float)(w0 + tid - 1 + k % 3);
            float y0f = floorf(py), x0f = floorf(px);
            int   y0 = (int)y0f,  x0 = (int)x0f;
            float ly = py - y0f,  lx = px - x0f;
            int   y1 = y0 + 1,    x1 = x0 + 1;
            float vy0 = (y0 >= 0 && y0 < Hdim) ? 1.f : 0.f;
            float vy1 = (y1 >= 0 && y1 < Hdim) ? 1.f : 0.f;
            float vx0 = (x0 >= 0 && x0 < Wdim) ? 1.f : 0.f;
            float vx1 = (x1 >= 0 && x1 < Wdim) ? 1.f : 0.f;
            int y0c = min(max(y0, 0), Hdim - 1), y1c = min(max(y1, 0), Hdim - 1);
            int x0c = min(max(x0, 0), Wdim - 1), x1c = min(max(x1, 0), Wdim - 1);
            s_off[tid][0] = y0c * Wdim + x0c;
            s_off[tid][1] = y0c * Wdim + x1c;
            s_off[tid][2] = y1c * Wdim + x0c;
            s_off[tid][3] = y1c * Wdim + x1c;
            s_wt[tid][0] = (1.f - ly) * (1.f - lx) * vy0 * vx0;
            s_wt[tid][1] = (1.f - ly) * lx         * vy0 * vx1;
            s_wt[tid][2] = ly * (1.f - lx)         * vy1 * vx0;
            s_wt[tid][3] = ly * lx                 * vy1 * vx1;
            s_m[tid] = msk[idx];
        }
        __syncthreads();
        {   // bilinear gather: each thread does 16 channels of one pixel;
            // consecutive lanes = consecutive pixels -> corner addresses land
            // in neighboring cache lines (x fully L2-resident: 33.5MB < 192MB)
            const int pi = tid & 31;
            const int c0 = (tid >> 5) << 4;
            const int o00 = s_off[pi][0], o01 = s_off[pi][1];
            const int o10 = s_off[pi][2], o11 = s_off[pi][3];
            const float w00 = s_wt[pi][0], w01 = s_wt[pi][1];
            const float w10 = s_wt[pi][2], w11 = s_wt[pi][3];
            const float mv  = s_m[pi];
            const float* xc = xb + (size_t)c0 * HW;
            #pragma unroll
            for (int cq = 0; cq < 16; ++cq) {
                const float* xp = xc + (size_t)cq * HW;
                float v = w00 * xp[o00] + w01 * xp[o01]
                        + w10 * xp[o10] + w11 * xp[o11];
                As[pi][c0 + cq] = (_Float16)(v * mv);
            }
        }
        __syncthreads();
        // 4 K-chunks of 32; B fragment loaded once, used by both M sub-tiles
        #pragma unroll
        for (int chunk = 0; chunk < 4; ++chunk) {
            const int cb = chunk << 5;
            const _Float16* bp =
                wp + ((((size_t)(k * 4 + chunk) * 8 + wv) * 32 + lane) << 4);
            v16h bf = *(const v16h*)bp;                          // 32B global load

            v8h a0lo = *(const v8h*)&As[mrow][cb + kofs];        // ds_load_b128
            v8h a0hi = *(const v8h*)&As[mrow][cb + kofs + 8];
            v16h a0 = __builtin_shufflevector(a0lo, a0hi,
                        0,1,2,3,4,5,6,7,8,9,10,11,12,13,14,15);
            acc0 = __builtin_amdgcn_wmma_f32_16x16x32_f16(
                       false, a0, false, bf, (short)0, acc0, false, false);

            v8h a1lo = *(const v8h*)&As[16 + mrow][cb + kofs];
            v8h a1hi = *(const v8h*)&As[16 + mrow][cb + kofs + 8];
            v16h a1 = __builtin_shufflevector(a1lo, a1hi,
                        0,1,2,3,4,5,6,7,8,9,10,11,12,13,14,15);
            acc1 = __builtin_amdgcn_wmma_f32_16x16x32_f16(
                       false, a1, false, bf, (short)0, acc1, false, false);
        }
    }

    // C/D layout: VGPR r = row M (pixel), lane = col N (out channel).
    // Each lane stores 8 consecutive pixels for one output channel, twice.
    const int n  = wv * 16 + mrow;
    const int m0 = (lane < 16) ? 0 : 8;
    float* op0 = out + (((size_t)(b * COUT + n)) << 14) + p_base + m0;
    float* op1 = op0 + 16;
    #pragma unroll
    for (int r = 0; r < 8; ++r) op0[r] = acc0[r];
    #pragma unroll
    for (int r = 0; r < 8; ++r) op1[r] = acc1[r];
}

// ---------------------------------------------------------------------------
extern "C" void kernel_launch(void* const* d_in, const int* in_sizes, int n_in,
                              void* d_out, int out_size, void* d_ws, size_t ws_size,
                              hipStream_t stream) {
    const float* x  = (const float*)d_in[0];   // (4,128,128,128)
    const float* ow = (const float*)d_in[1];   // (27,128,3,3)
    const float* ob = (const float*)d_in[2];   // (27,)
    const float* wg = (const float*)d_in[3];   // (128,128,3,3)
    float* out = (float*)d_out;

    const size_t S = (size_t)Bsz * KK * HW;    // 589824 floats per array
    float* dy  = (float*)d_ws;
    float* dx  = dy + S;
    float* msk = dx + S;
    _Float16* wp = (_Float16*)(msk + S);       // 147456 halves (288 KB)

    pack_weight_kernel<<<(KK * 4 * 8 * 32 * 16 + 255) / 256, 256, 0, stream>>>(wg, wp);
    offset_conv_kernel<<<(Bsz * HW) / 256, 256, 0, stream>>>(x, ow, ob, dy, dx, msk);
    deform_wmma_kernel<<<Bsz * (HW / TILE_M), 256, 0, stream>>>(x, wp, dy, dx, msk, out);
}
